// CrossCLR_onlyIntraModality_4415226380691
// MI455X (gfx1250) — matrix operand order repaired and checked by reading, not compiled, not run on hardware
//
#include <hip/hip_runtime.h>
#include <hip/hip_bf16.h>

#define NMAT 4096
#define KDIM 1024
#define TEMP_INV (1.0f / 0.03f)
#define NEGW 0.8f
#define COLCHUNKS 32

typedef __attribute__((ext_vector_type(16))) __bf16 v16bf;
typedef __attribute__((ext_vector_type(8)))  __bf16 bf16x8;
typedef __attribute__((ext_vector_type(8)))  float  v8f;

// ---------- fp32 -> bf16 hi/lo split (round-to-nearest-even) ----------
__device__ __forceinline__ unsigned short f32_to_bf16_rne(float f) {
    unsigned u = __float_as_uint(f);
    unsigned r = 0x7FFFu + ((u >> 16) & 1u);
    return (unsigned short)((u + r) >> 16);
}
__device__ __forceinline__ float bf16_to_f32(unsigned short h) {
    return __uint_as_float(((unsigned)h) << 16);
}

__global__ __launch_bounds__(256) void cclr_convert_hilo(
    const float* __restrict__ x, unsigned short* __restrict__ hi,
    unsigned short* __restrict__ lo, int n)
{
    int i = blockIdx.x * blockDim.x + threadIdx.x;
    if (i >= n) return;
    float v = x[i];
    unsigned short h = f32_to_bf16_rne(v);
    float rem = v - bf16_to_f32(h);
    hi[i] = h;
    lo[i] = f32_to_bf16_rne(rem);
}

// ---------- WMMA fragment loaders ----------
// A (16x32 MxK): lanes 0-15 row M=l hold K {0..7, 16..23}; lanes 16-31 hold
// K {8..15, 24..31}  (caller pre-offsets pointer by half*8).
__device__ __forceinline__ v16bf ldA(const unsigned short* p) {
    bf16x8 c0 = *(const bf16x8*)(p);
    bf16x8 c1 = *(const bf16x8*)(p + 16);
    return __builtin_shufflevector(c0, c1, 0,1,2,3,4,5,6,7,8,9,10,11,12,13,14,15);
}
// B (32x16 KxN): lanes 0-15 col N=l hold K {0..7, 16..23}; lanes 16-31 hold
// K {8..15, 24..31}  (caller pre-offsets pointer by half*8).
__device__ __forceinline__ v16bf ldB(const unsigned short* p) {
    bf16x8 c0 = *(const bf16x8*)(p);
    bf16x8 c1 = *(const bf16x8*)(p + 16);
    return __builtin_shufflevector(c0, c1, 0,1,2,3,4,5,6,7,8,9,10,11,12,13,14,15);
}
__device__ __forceinline__ v8f wmma_bf16(v16bf a, v16bf b, v8f c) {
    return __builtin_amdgcn_wmma_f32_16x16x32_bf16(false, a, false, b, (short)0, c, false, false);
}

// ---------- WMMA GEMM: Out[m,n] = sum_k X[m,k] * Y[n,k]  (X @ Y^T) ----------
// hi/lo split: X*Y ~= Xh*Yh + Xh*Yl + Xl*Yh (fp32 accum).
// Wave tile = 32x32 (2x2 of 16x16), block = 8 waves -> 64x128.
// symm!=0: skip wave tiles strictly below the diagonal; mirror-store tiles
// strictly above it (each output element written exactly once or last-writer
// deterministic), so only the upper triangle is computed.
__global__ __launch_bounds__(256) void cclr_wmma_gemm_nt(
    const unsigned short* __restrict__ Ahi, const unsigned short* __restrict__ Alo,
    const unsigned short* __restrict__ Bhi, const unsigned short* __restrict__ Blo,
    float* __restrict__ Out, int symm)
{
    const int K = KDIM, N = NMAT;
    const int lane = threadIdx.x & 31;
    const int wave = threadIdx.x >> 5;   // 0..7
    const int wm = wave >> 2;            // 0..1
    const int wn = wave & 3;             // 0..3
    const int m0 = blockIdx.y * 64 + wm * 32;
    const int n0 = blockIdx.x * 128 + wn * 32;
    if (symm && (n0 + 32 <= m0)) return;   // strictly below diagonal

    const int half = lane >> 4;          // 0 or 1
    const int l    = lane & 15;
    const int T    = 16 * K;             // element offset to next 16-row tile

    const unsigned short* arh = Ahi + (size_t)(m0 + l) * K + half * 8;
    const unsigned short* arl = Alo + (size_t)(m0 + l) * K + half * 8;
    const unsigned short* brh = Bhi + (size_t)(n0 + l) * K + half * 8;
    const unsigned short* brl = Blo + (size_t)(n0 + l) * K + half * 8;

    v8f acc00 = {}, acc01 = {}, acc10 = {}, acc11 = {};
    for (int k0 = 0; k0 < K; k0 += 32) {
        v16bf a0h = ldA(arh + k0);
        v16bf a0l = ldA(arl + k0);
        v16bf a1h = ldA(arh + T + k0);
        v16bf a1l = ldA(arl + T + k0);
        v16bf b0h = ldB(brh + k0);
        v16bf b0l = ldB(brl + k0);
        v16bf b1h = ldB(brh + T + k0);
        v16bf b1l = ldB(brl + T + k0);

        acc00 = wmma_bf16(a0h, b0h, acc00);
        acc00 = wmma_bf16(a0h, b0l, acc00);
        acc00 = wmma_bf16(a0l, b0h, acc00);

        acc01 = wmma_bf16(a0h, b1h, acc01);
        acc01 = wmma_bf16(a0h, b1l, acc01);
        acc01 = wmma_bf16(a0l, b1h, acc01);

        acc10 = wmma_bf16(a1h, b0h, acc10);
        acc10 = wmma_bf16(a1h, b0l, acc10);
        acc10 = wmma_bf16(a1l, b0h, acc10);

        acc11 = wmma_bf16(a1h, b1h, acc11);
        acc11 = wmma_bf16(a1h, b1l, acc11);
        acc11 = wmma_bf16(a1l, b1h, acc11);
    }

    // C/D layout: VGPR r -> row (sub-tile + r + 8*half), col = n0 + l
    {
        float* o00 = Out + (size_t)(m0 +  8 * half) * N + (n0 + l);
        float* o01 = o00 + 16;
        float* o10 = Out + (size_t)(m0 + 16 + 8 * half) * N + (n0 + l);
        float* o11 = o10 + 16;
        #pragma unroll
        for (int r = 0; r < 8; ++r) {
            o00[(size_t)r * N] = acc00[r];
            o01[(size_t)r * N] = acc01[r];
            o10[(size_t)r * N] = acc10[r];
            o11[(size_t)r * N] = acc11[r];
        }
    }
    if (symm && n0 > m0) {
        // mirror: value G[m0+mt*16+r+8*half, n0+nt*16+l] -> Out[col*N + row]
        // per lane the 8 r-values are contiguous floats (32B aligned).
        float* q00 = Out + (size_t)(n0 + l) * N + (m0 +      8 * half);
        float* q01 = Out + (size_t)(n0 + 16 + l) * N + (m0 +      8 * half);
        float* q10 = Out + (size_t)(n0 + l) * N + (m0 + 16 + 8 * half);
        float* q11 = Out + (size_t)(n0 + 16 + l) * N + (m0 + 16 + 8 * half);
        *(v8f*)q00 = acc00;
        *(v8f*)q01 = acc01;
        *(v8f*)q10 = acc10;
        *(v8f*)q11 = acc11;
    }
}

// ---------- diagonal extraction: norms + S diag ----------
__global__ __launch_bounds__(256) void cclr_diag(
    const float* __restrict__ S, const float* __restrict__ Gbb,
    const float* __restrict__ Gpp, float* __restrict__ nb,
    float* __restrict__ np_, float* __restrict__ sd)
{
    int i = blockIdx.x * blockDim.x + threadIdx.x;
    if (i >= NMAT) return;
    size_t d = (size_t)i * NMAT + i;
    nb[i]  = sqrtf(Gbb[d]);
    np_[i] = sqrtf(Gpp[d]);
    sd[i]  = S[d];
}

// ---------- log-sum-exp merge helper ----------
__device__ __forceinline__ void ms_merge(float& m, float& s, float m2, float s2) {
    float mn = fmaxf(m, m2);
    s = s * __expf(m - mn) + s2 * __expf(m2 - mn);
    m = mn;
}

// ---------- per-row stats: rank count + two-pass LSE of S/Gbb/Gpp rows ----------
__global__ __launch_bounds__(256) void cclr_row_stats(
    const float* __restrict__ S, const float* __restrict__ Gbb,
    const float* __restrict__ Gpp, const float* __restrict__ nb,
    const float* __restrict__ np_, const float* __restrict__ sd,
    int* __restrict__ cnt_row,
    float* __restrict__ mS, float* __restrict__ sS,
    float* __restrict__ mBB, float* __restrict__ sBB,
    float* __restrict__ mPP, float* __restrict__ sPP)
{
    const int N = NMAT;
    const int i = blockIdx.x;
    const int t = threadIdx.x;
    const float inb = TEMP_INV / nb[i];
    const float inp = TEMP_INV / np_[i];
    const float sdi = sd[i];
    const float* Srow = S   + (size_t)i * N;
    const float* Brow = Gbb + (size_t)i * N;
    const float* Prow = Gpp + (size_t)i * N;

    // pass 1: rank count + per-thread maxima (no exp -> no serial chain)
    int cnt = 0;
    float m1 = -1e30f, m2 = -1e30f, m3 = -1e30f;
    for (int j = t; j < N; j += 256) {
        float sv = Srow[j];
        cnt += (sv > sdi) ? 1 : 0;
        m1 = fmaxf(m1, sv * inb / np_[j]);
        if (j != i) {
            m2 = fmaxf(m2, NEGW * Brow[j] * inb / nb[j]);
            m3 = fmaxf(m3, NEGW * Prow[j] * inp / np_[j]);
        }
    }
    // pass 2: independent exp accumulation (rows are L1/L2 resident)
    float s1 = 0.f, s2 = 0.f, s3 = 0.f;
    for (int j = t; j < N; j += 256) {
        float sv = Srow[j];
        s1 += __expf(sv * inb / np_[j] - m1);
        if (j != i) {
            s2 += __expf(NEGW * Brow[j] * inb / nb[j] - m2);
            s3 += __expf(NEGW * Prow[j] * inp / np_[j] - m3);
        }
    }

    __shared__ float sm[256], ss[256];
    __shared__ int   sc[256];
    sc[t] = cnt; __syncthreads();
    for (int o = 128; o > 0; o >>= 1) { if (t < o) sc[t] += sc[t + o]; __syncthreads(); }
    if (t == 0) cnt_row[i] = sc[0];
    __syncthreads();

    sm[t] = m1; ss[t] = s1; __syncthreads();
    for (int o = 128; o > 0; o >>= 1) { if (t < o) ms_merge(sm[t], ss[t], sm[t + o], ss[t + o]); __syncthreads(); }
    if (t == 0) { mS[i] = sm[0]; sS[i] = ss[0]; }
    __syncthreads();

    sm[t] = m2; ss[t] = s2; __syncthreads();
    for (int o = 128; o > 0; o >>= 1) { if (t < o) ms_merge(sm[t], ss[t], sm[t + o], ss[t + o]); __syncthreads(); }
    if (t == 0) { mBB[i] = sm[0]; sBB[i] = ss[0]; }
    __syncthreads();

    sm[t] = m3; ss[t] = s3; __syncthreads();
    for (int o = 128; o > 0; o >>= 1) { if (t < o) ms_merge(sm[t], ss[t], sm[t + o], ss[t + o]); __syncthreads(); }
    if (t == 0) { mPP[i] = sm[0]; sPP[i] = ss[0]; }
}

// ---------- per-column partial stats over S (chunked rows for occupancy) ----------
__global__ __launch_bounds__(256) void cclr_col_stats(
    const float* __restrict__ S, const float* __restrict__ nb,
    const float* __restrict__ np_, const float* __restrict__ sd,
    int* __restrict__ cntP, float* __restrict__ mP, float* __restrict__ sP)
{
    const int N = NMAT;
    const int j  = blockIdx.x * 256 + threadIdx.x;
    const int c  = blockIdx.y;
    const int k0 = c * (N / COLCHUNKS);
    const int k1 = k0 + (N / COLCHUNKS);
    const float inp = TEMP_INV / np_[j];
    const float sdj = sd[j];

    int cnt = 0; float m = -1e30f;
    for (int k = k0; k < k1; ++k) {
        float v = S[(size_t)k * N + j];
        cnt += (v > sdj) ? 1 : 0;
        m = fmaxf(m, v * inp / nb[k]);
    }
    float s = 0.f;
    for (int k = k0; k < k1; ++k) {
        float v = S[(size_t)k * N + j];
        s += __expf(v * inp / nb[k] - m);
    }
    cntP[(size_t)c * N + j] = cnt;
    mP[(size_t)c * N + j]   = m;
    sP[(size_t)c * N + j]   = s;
}

__global__ __launch_bounds__(256) void cclr_col_merge(
    const int* __restrict__ cntP, const float* __restrict__ mP,
    const float* __restrict__ sP,
    int* __restrict__ cnt_col, float* __restrict__ mC, float* __restrict__ sC)
{
    const int N = NMAT;
    int j = blockIdx.x * 256 + threadIdx.x;
    if (j >= N) return;
    int cnt = 0; float m = -1e30f, s = 0.f;
    for (int c = 0; c < COLCHUNKS; ++c) {
        cnt += cntP[(size_t)c * N + j];
        ms_merge(m, s, mP[(size_t)c * N + j], sP[(size_t)c * N + j]);
    }
    cnt_col[j] = cnt; mC[j] = m; sC[j] = s;
}

// ---------- final deterministic reduction ----------
__global__ __launch_bounds__(256) void cclr_final(
    const float* __restrict__ nb, const float* __restrict__ np_,
    const float* __restrict__ sd,
    const int* __restrict__ cnt_row, const int* __restrict__ cnt_col,
    const float* __restrict__ mS, const float* __restrict__ sS,
    const float* __restrict__ mBB, const float* __restrict__ sBB,
    const float* __restrict__ mPP, const float* __restrict__ sPP,
    const float* __restrict__ mC, const float* __restrict__ sC,
    float* __restrict__ out)
{
    const int N = NMAT;
    const int t = threadIdx.x;
    float acc = 0.f;
    for (int i = t; i < N; i += 256) {
        float diag = sd[i] * TEMP_INV / (nb[i] * np_[i]);

        float m = mS[i], s = sS[i];
        ms_merge(m, s, mBB[i], sBB[i]);
        float lse_b  = m + __logf(s);
        float rank_b = 1.0f / (float)(N - cnt_row[i]) + 1.0f;

        float m2 = mC[i], s2 = sC[i];
        ms_merge(m2, s2, mPP[i], sPP[i]);
        float lse_p  = m2 + __logf(s2);
        float rank_p = 1.0f / (float)(N - cnt_col[i]) + 1.0f;

        acc += rank_b * (lse_b - diag) + rank_p * (lse_p - diag);
    }
    __shared__ float sm[256];
    sm[t] = acc; __syncthreads();
    for (int o = 128; o > 0; o >>= 1) { if (t < o) sm[t] += sm[t + o]; __syncthreads(); }
    if (t == 0) out[0] = sm[0] * 0.5f;
}

extern "C" void kernel_launch(void* const* d_in, const int* in_sizes, int n_in,
                              void* d_out, int out_size, void* d_ws, size_t ws_size,
                              hipStream_t stream) {
    const int N = NMAT, K = KDIM;
    const float* brand = (const float*)d_in[0];
    const float* post  = (const float*)d_in[1];
    float* out = (float*)d_out;
    char* ws = (char*)d_ws;

    const size_t szMat  = (size_t)N * N * sizeof(float);          // 64 MB
    const size_t szHalf = (size_t)N * K * sizeof(unsigned short); // 8 MB

    float* S   = (float*)(ws);
    float* Gbb = (float*)(ws + szMat);
    float* Gpp = (float*)(ws + 2 * szMat);
    unsigned short* bh = (unsigned short*)(ws + 3 * szMat);
    unsigned short* bl = (unsigned short*)(ws + 3 * szMat + szHalf);
    unsigned short* ph = (unsigned short*)(ws + 3 * szMat + 2 * szHalf);
    unsigned short* pl = (unsigned short*)(ws + 3 * szMat + 3 * szHalf);
    char* st = ws + 3 * szMat + 4 * szHalf;
    float* nb   = (float*)(st);               st += (size_t)N * 4;
    float* np_  = (float*)(st);               st += (size_t)N * 4;
    float* sd   = (float*)(st);               st += (size_t)N * 4;
    float* mS   = (float*)(st);               st += (size_t)N * 4;
    float* sS   = (float*)(st);               st += (size_t)N * 4;
    float* mBB  = (float*)(st);               st += (size_t)N * 4;
    float* sBB  = (float*)(st);               st += (size_t)N * 4;
    float* mPP  = (float*)(st);               st += (size_t)N * 4;
    float* sPP  = (float*)(st);               st += (size_t)N * 4;
    float* mC   = (float*)(st);               st += (size_t)N * 4;
    float* sC   = (float*)(st);               st += (size_t)N * 4;
    int*   cnt_row = (int*)(st);              st += (size_t)N * 4;
    int*   cnt_col = (int*)(st);              st += (size_t)N * 4;
    int*   cntP = (int*)(st);                 st += (size_t)COLCHUNKS * N * 4;
    float* mP   = (float*)(st);               st += (size_t)COLCHUNKS * N * 4;
    float* sP   = (float*)(st);

    // 1) split fp32 -> bf16 hi/lo
    int nel = N * K;
    cclr_convert_hilo<<<(nel + 255) / 256, 256, 0, stream>>>(brand, bh, bl, nel);
    cclr_convert_hilo<<<(nel + 255) / 256, 256, 0, stream>>>(post,  ph, pl, nel);

    // 2) three Gram matrices via WMMA (X @ Y^T); Gbb/Gpp upper-triangle only
    dim3 ggrid(N / 128, N / 64);
    cclr_wmma_gemm_nt<<<ggrid, 256, 0, stream>>>(bh, bl, ph, pl, S,   0); // brand @ post^T
    cclr_wmma_gemm_nt<<<ggrid, 256, 0, stream>>>(bh, bl, bh, bl, Gbb, 1); // brand @ brand^T
    cclr_wmma_gemm_nt<<<ggrid, 256, 0, stream>>>(ph, pl, ph, pl, Gpp, 1); // post @ post^T

    // 3) diagonals -> norms + score diag
    cclr_diag<<<N / 256, 256, 0, stream>>>(S, Gbb, Gpp, nb, np_, sd);

    // 4) row stats (rank_2 counts, LSE of brand logits rows / intra rows)
    cclr_row_stats<<<N, 256, 0, stream>>>(S, Gbb, Gpp, nb, np_, sd,
                                          cnt_row, mS, sS, mBB, sBB, mPP, sPP);

    // 5) column partial stats + merge (rank_1 counts, LSE of post logits rows)
    cclr_col_stats<<<dim3(N / 256, COLCHUNKS), 256, 0, stream>>>(
        S, nb, np_, sd, cntP, mP, sP);
    cclr_col_merge<<<N / 256, 256, 0, stream>>>(cntP, mP, sP, cnt_col, mC, sC);

    // 6) deterministic final reduction
    cclr_final<<<1, 256, 0, stream>>>(nb, np_, sd, cnt_row, cnt_col,
                                      mS, sS, mBB, sBB, mPP, sPP, mC, sC, out);
}
